// DecoderLSTM_51273319580055
// MI455X (gfx1250) — matrix operand order repaired and checked by reading, not compiled, not run on hardware
//
#include <hip/hip_runtime.h>
#include <hip/hip_bf16.h>

// ---- dims ----
#define Bdim 256
#define Sdim 512
#define Hdim 512
#define Edim 256
#define Vdim 16000
#define KXH  1280   // E + H + H  (emb | context | h)
#define G4H  2048   // 4*H

typedef __bf16 bf16_t;
typedef __attribute__((ext_vector_type(16))) __bf16 v16bf;
typedef __attribute__((ext_vector_type(8)))  __bf16 v8bf;
typedef __attribute__((ext_vector_type(8)))  float  v8f;

// Build A fragment for v_wmma_f32_16x16x32_bf16.
// Lane layout (contiguous, matching B): lanes 0-15 row M=lane hold K {0..15},
// lanes 16-31 row M=lane-16 hold K {16..31}. Caller passes kb = ktile*32 + (lane>=16 ? 16 : 0).
__device__ __forceinline__ v16bf load_a_frag(const bf16_t* row, int kb) {
    v8bf lo = *(const v8bf*)(row + kb);        // K kb..kb+7
    v8bf hi = *(const v8bf*)(row + kb + 8);    // K kb+8..kb+15
    v16bf a;
#pragma unroll
    for (int i = 0; i < 8; ++i) { a[i] = lo[i]; a[i + 8] = hi[i]; }
    return a;
}

__device__ __forceinline__ v8f wmma_bf16(v16bf a, v16bf b, v8f c) {
    return __builtin_amdgcn_wmma_f32_16x16x32_bf16(false, a, false, b, (short)0, c, false, false);
}

// ---------------- prep / conversion kernels ----------------

__global__ void k_cvt(const float* __restrict__ src, bf16_t* __restrict__ dst, int n) {
    int i = blockIdx.x * 256 + threadIdx.x;
    if (i < n) dst[i] = (bf16_t)src[i];
}

// We[n][k] = attn_W[n][H + k]  (enc-half of attn_W, n-major bf16: native B-fragment layout)
__global__ void k_build_we(const float* __restrict__ attn_W, bf16_t* __restrict__ We) {
    int i = blockIdx.x * 256 + threadIdx.x;          // 512*512
    int n = i >> 9, k = i & 511;
    We[i] = (bf16_t)attn_W[n * 1024 + 512 + k];
}

// Wcat[n][k] = k<768 ? W_ih[n][k] : W_hh[n][k-768]   (n-major bf16)
__global__ void k_build_wcat(const float* __restrict__ W_ih, const float* __restrict__ W_hh,
                             bf16_t* __restrict__ W) {
    int i = blockIdx.x * 256 + threadIdx.x;          // 2048*1280
    int n = i / KXH, k = i % KXH;
    float v = (k < 768) ? W_ih[n * 768 + k] : W_hh[n * 512 + (k - 768)];
    W[i] = (bf16_t)v;
}

// xh[b][0..255]=emb(tgt[b]), xh[b][768..1279]=h[b]  (context slot 256..767 filled later)
__global__ void k_xh_init(const int* __restrict__ tgt, const float* __restrict__ emb,
                          const float* __restrict__ h, bf16_t* __restrict__ xh) {
    int i = blockIdx.x * 256 + threadIdx.x;          // 256*768
    int b = i / 768, q = i % 768;
    float v; int dst;
    if (q < 256) { v = emb[(size_t)tgt[b] * Edim + q]; dst = b * KXH + q; }
    else         { v = h[b * Hdim + (q - 256)];        dst = b * KXH + 512 + q; }
    xh[dst] = (bf16_t)v;
}

__global__ void k_bias(const float* __restrict__ bih, const float* __restrict__ bhh,
                       float* __restrict__ bias) {
    int i = blockIdx.x * 256 + threadIdx.x;
    if (i < G4H) bias[i] = bih[i] + bhh[i];
}

// ph[b][j] = attn_b[j] + sum_k h[b][k] * attn_W[j][k]   (h-half, hoisted out of the S loop)
__global__ void k_ph(const float* __restrict__ h, const float* __restrict__ attn_W,
                     const float* __restrict__ attn_b, float* __restrict__ ph) {
    int idx = blockIdx.x * 256 + threadIdx.x;        // B*H
    int b = idx >> 9, j = idx & 511;
    const float4* hv = (const float4*)(h + b * Hdim);
    const float4* wv = (const float4*)(attn_W + (size_t)j * 1024);
    float acc = attn_b[j];
#pragma unroll 4
    for (int k = 0; k < 128; ++k) {
        float4 a = hv[k], w = wv[k];
        acc += a.x * w.x + a.y * w.y + a.z * w.z + a.w * w.w;
    }
    ph[idx] = acc;
}

// ---------------- attention scores: fused GEMM + tanh + v-reduction (WMMA) ----------------
// grid (S/64, B), block 128 (4 waves). Wave w owns 16-row m-tile w; loops all 512 N in groups of 4 tiles.
__global__ void k_attn_scores(const float* __restrict__ enc, const bf16_t* __restrict__ We,
                              const float* __restrict__ ph, const float* __restrict__ vvec,
                              float* __restrict__ scores) {
    __shared__ __align__(16) bf16_t Als[64][520];    // 64 s-rows x 512 K, +8 halves pad (16B) per row
    int b = blockIdx.y, s0 = blockIdx.x * 64;
    int tid = threadIdx.x;
    const float* eb = enc + ((size_t)b * Sdim + s0) * Hdim;
    // stage + fp32->bf16 convert 64x512 tile
    for (int i = tid; i < 64 * 128; i += 128) {
        int row = i >> 7, kq = (i & 127) << 2;
        float4 f = *(const float4*)(eb + row * Hdim + kq);
        Als[row][kq + 0] = (bf16_t)f.x; Als[row][kq + 1] = (bf16_t)f.y;
        Als[row][kq + 2] = (bf16_t)f.z; Als[row][kq + 3] = (bf16_t)f.w;
    }
    __syncthreads();

    int lane = tid & 31, wave = tid >> 5;
    const bf16_t* arow = &Als[wave * 16 + (lane & 15)][0];
    const int kselA = (lane >> 4) * 16;
    const int kselB = (lane >> 4) * 16;
    const float* phb = ph + b * Hdim;

    float ps[8];
#pragma unroll
    for (int r = 0; r < 8; ++r) ps[r] = 0.f;

    for (int ng = 0; ng < 8; ++ng) {                 // 8 groups x 4 n-tiles = 512 N
        v8f acc[4];
#pragma unroll
        for (int j = 0; j < 4; ++j)
#pragma unroll
            for (int r = 0; r < 8; ++r) acc[j][r] = 0.f;
        int nn[4];
#pragma unroll
        for (int j = 0; j < 4; ++j) nn[j] = ng * 64 + j * 16 + (lane & 15);

        for (int kt = 0; kt < 16; ++kt) {            // K = 512
            v16bf a = load_a_frag(arow, kt * 32 + kselA);
#pragma unroll
            for (int j = 0; j < 4; ++j) {
                v16bf bb = *(const v16bf*)(We + (size_t)nn[j] * 512 + kt * 32 + kselB);
                acc[j] = wmma_bf16(a, bb, acc[j]);
            }
        }
        // epilogue: scores += v[n] * tanh(gemm + ph[b][n])  (ph already includes attn_b)
#pragma unroll
        for (int j = 0; j < 4; ++j) {
            float phn = phb[nn[j]], vn = vvec[nn[j]];
#pragma unroll
            for (int r = 0; r < 8; ++r) ps[r] += vn * tanhf(acc[j][r] + phn);
        }
    }
    // reduce over the 16 lanes of each half (N columns); rows r / r+8 per half
#pragma unroll
    for (int off = 8; off >= 1; off >>= 1)
#pragma unroll
        for (int r = 0; r < 8; ++r) ps[r] += __shfl_xor(ps[r], off, 32);
    if ((lane & 15) == 0) {
        int mrow = s0 + wave * 16 + ((lane >> 4) ? 8 : 0);
#pragma unroll
        for (int r = 0; r < 8; ++r) scores[b * Sdim + mrow + r] = ps[r];
    }
}

// ---------------- softmax over S per batch row ----------------
__global__ void k_softmax(const float* __restrict__ scores, float* __restrict__ attn) {
    int b = blockIdx.x, t = threadIdx.x;             // 256 threads, 2 elems each
    __shared__ float red[256];
    float v0 = scores[b * Sdim + t], v1 = scores[b * Sdim + 256 + t];
    red[t] = fmaxf(v0, v1); __syncthreads();
    for (int s = 128; s > 0; s >>= 1) { if (t < s) red[t] = fmaxf(red[t], red[t + s]); __syncthreads(); }
    float mx = red[0]; __syncthreads();
    float e0 = __expf(v0 - mx), e1 = __expf(v1 - mx);
    red[t] = e0 + e1; __syncthreads();
    for (int s = 128; s > 0; s >>= 1) { if (t < s) red[t] += red[t + s]; __syncthreads(); }
    float inv = 1.0f / red[0];
    attn[b * Sdim + t] = e0 * inv;
    attn[b * Sdim + 256 + t] = e1 * inv;
}

// ---------------- context = attn @ enc, written straight into bf16 xh slot ----------------
__global__ void k_context(const float* __restrict__ attn, const float* __restrict__ enc,
                          bf16_t* __restrict__ xh) {
    int b = blockIdx.x, h = threadIdx.x;             // 512 threads
    __shared__ float aw[512];
    aw[h] = attn[b * Sdim + h];
    __syncthreads();
    const float* eb = enc + (size_t)b * Sdim * Hdim;
    float acc = 0.f;
#pragma unroll 4
    for (int s = 0; s < Sdim; ++s) acc += aw[s] * eb[s * Hdim + h];
    xh[b * KXH + 256 + h] = (bf16_t)acc;
}

// ---------------- LSTM gates GEMM (WMMA): M=256, N=2048, K=1280 ----------------
// grid (N/64, M/64), block 128 (4 waves); wave = m-tile, 4 n-tiles, K-loop 40.
__global__ void k_gates(const bf16_t* __restrict__ xh, const bf16_t* __restrict__ W,
                        const float* __restrict__ bias, float* __restrict__ gates) {
    int n0 = blockIdx.x * 64, m0 = blockIdx.y * 64;
    int lane = threadIdx.x & 31, wave = threadIdx.x >> 5;
    const bf16_t* arow = xh + (size_t)(m0 + wave * 16 + (lane & 15)) * KXH;
    const int kselA = (lane >> 4) * 16, kselB = (lane >> 4) * 16;

    v8f acc[4];
    int nn[4]; const bf16_t* brow[4];
#pragma unroll
    for (int j = 0; j < 4; ++j) {
        nn[j] = n0 + j * 16 + (lane & 15);
        brow[j] = W + (size_t)nn[j] * KXH;
#pragma unroll
        for (int r = 0; r < 8; ++r) acc[j][r] = 0.f;
    }
    for (int kt = 0; kt < 40; ++kt) {
        v16bf a = load_a_frag(arow, kt * 32 + kselA);
#pragma unroll
        for (int j = 0; j < 4; ++j) {
            v16bf bb = *(const v16bf*)(brow[j] + kt * 32 + kselB);
            acc[j] = wmma_bf16(a, bb, acc[j]);
        }
    }
    int rbase = m0 + wave * 16 + ((lane >> 4) ? 8 : 0);
#pragma unroll
    for (int j = 0; j < 4; ++j) {
        float bs = bias[nn[j]];
#pragma unroll
        for (int r = 0; r < 8; ++r) gates[(size_t)(rbase + r) * G4H + nn[j]] = acc[j][r] + bs;
    }
}

// ---------------- LSTM elementwise ----------------
__global__ void k_lstm(const float* __restrict__ gates, const float* __restrict__ cell,
                       float* __restrict__ hnew, float* __restrict__ cnew,
                       bf16_t* __restrict__ hnbf) {
    int idx = blockIdx.x * 256 + threadIdx.x;        // B*H
    int b = idx >> 9, h = idx & 511;
    const float* g = gates + (size_t)b * G4H;
    float gi = g[h], gf = g[512 + h], gg = g[1024 + h], go = g[1536 + h];
    float si = 1.f / (1.f + __expf(-gi));
    float sf = 1.f / (1.f + __expf(-gf));
    float so = 1.f / (1.f + __expf(-go));
    float cn = sf * cell[idx] + si * tanhf(gg);
    float hn = so * tanhf(cn);
    cnew[idx] = cn; hnew[idx] = hn; hnbf[idx] = (bf16_t)hn;
}

// ---------------- prediction = h_new @ fc_W^T + fc_b (WMMA): M=256, N=16000, K=512 ----------------
__global__ void k_pred(const bf16_t* __restrict__ hn, const bf16_t* __restrict__ W,
                       const float* __restrict__ fcb, float* __restrict__ pred) {
    int n0 = blockIdx.x * 64, m0 = blockIdx.y * 64;
    int lane = threadIdx.x & 31, wave = threadIdx.x >> 5;
    const bf16_t* arow = hn + (size_t)(m0 + wave * 16 + (lane & 15)) * Hdim;
    const int kselA = (lane >> 4) * 16, kselB = (lane >> 4) * 16;

    v8f acc[4];
    int nn[4]; const bf16_t* brow[4];
#pragma unroll
    for (int j = 0; j < 4; ++j) {
        nn[j] = n0 + j * 16 + (lane & 15);
        brow[j] = W + (size_t)nn[j] * Hdim;
#pragma unroll
        for (int r = 0; r < 8; ++r) acc[j][r] = 0.f;
    }
    for (int kt = 0; kt < 16; ++kt) {
        v16bf a = load_a_frag(arow, kt * 32 + kselA);
#pragma unroll
        for (int j = 0; j < 4; ++j) {
            __builtin_prefetch(brow[j] + (kt + 1) * 32 + kselB, 0, 1);
            v16bf bb = *(const v16bf*)(brow[j] + kt * 32 + kselB);
            acc[j] = wmma_bf16(a, bb, acc[j]);
        }
    }
    int rbase = m0 + wave * 16 + ((lane >> 4) ? 8 : 0);
#pragma unroll
    for (int j = 0; j < 4; ++j) {
        float bs = fcb[nn[j]];
#pragma unroll
        for (int r = 0; r < 8; ++r) pred[(size_t)(rbase + r) * Vdim + nn[j]] = acc[j][r] + bs;
    }
}

// ---------------- host launcher ----------------
extern "C" void kernel_launch(void* const* d_in, const int* in_sizes, int n_in,
                              void* d_out, int out_size, void* d_ws, size_t ws_size,
                              hipStream_t stream) {
    const int*   tgt     = (const int*)  d_in[0];
    const float* hidden  = (const float*)d_in[1];
    const float* cell    = (const float*)d_in[2];
    const float* enc     = (const float*)d_in[3];
    const float* emb     = (const float*)d_in[4];
    const float* attn_W  = (const float*)d_in[5];
    const float* attn_b  = (const float*)d_in[6];
    const float* vvec    = (const float*)d_in[7];
    const float* W_ih    = (const float*)d_in[8];
    const float* W_hh    = (const float*)d_in[9];
    const float* b_ih    = (const float*)d_in[10];
    const float* b_hh    = (const float*)d_in[11];
    const float* fc_W    = (const float*)d_in[12];
    const float* fc_b    = (const float*)d_in[13];

    // outputs, concatenated: prediction | h_new | c_new | attn
    float* pred = (float*)d_out;
    float* hnew = pred + (size_t)Bdim * Vdim;
    float* cnew = hnew + (size_t)Bdim * Hdim;
    float* attn = cnew + (size_t)Bdim * Hdim;

    // workspace layout (~26 MB)
    char* w = (char*)d_ws;
    float*  ph      = (float*) (w);                           //  512 KB
    float*  scores  = (float*) (w + (size_t)( 512 << 10));    //  512 KB
    float*  biascat = (float*) (w + (size_t)(1024 << 10));    //    8 KB
    float*  gates   = (float*) (w + (size_t)(1056 << 10));    // 2048 KB
    bf16_t* We      = (bf16_t*)(w + (size_t)(3104 << 10));    //  512 KB
    bf16_t* xh      = (bf16_t*)(w + (size_t)(3616 << 10));    //  640 KB
    bf16_t* hnbf    = (bf16_t*)(w + (size_t)(4256 << 10));    //  256 KB
    bf16_t* wcat    = (bf16_t*)(w + (size_t)(4512 << 10));    // 5120 KB
    bf16_t* fcwb    = (bf16_t*)(w + (size_t)(9632 << 10));    // 16000 KB

    // --- prep / conversions (independent) ---
    k_cvt       <<<Vdim * Hdim / 256, 256, 0, stream>>>(fc_W, fcwb, Vdim * Hdim);
    k_build_we  <<<Hdim * Hdim / 256, 256, 0, stream>>>(attn_W, We);
    k_build_wcat<<<G4H * KXH / 256,   256, 0, stream>>>(W_ih, W_hh, wcat);
    k_xh_init   <<<Bdim * 768 / 256,  256, 0, stream>>>(tgt, emb, hidden, xh);
    k_bias      <<<G4H / 256,         256, 0, stream>>>(b_ih, b_hh, biascat);
    k_ph        <<<Bdim * Hdim / 256, 256, 0, stream>>>(hidden, attn_W, attn_b, ph);

    // --- attention ---
    k_attn_scores<<<dim3(Sdim / 64, Bdim), 128, 0, stream>>>(enc, We, ph, vvec, scores);
    k_softmax    <<<Bdim, 256, 0, stream>>>(scores, attn);
    k_context    <<<Bdim, 512, 0, stream>>>(attn, enc, xh);

    // --- LSTM cell ---
    k_gates<<<dim3(G4H / 64, Bdim / 64), 128, 0, stream>>>(xh, wcat, biascat, gates);
    k_lstm <<<Bdim * Hdim / 256, 256, 0, stream>>>(gates, cell, hnew, cnew, hnbf);

    // --- vocab projection ---
    k_pred<<<dim3(Vdim / 64, Bdim / 64), 128, 0, stream>>>(hnbf, fcwb, fc_b, pred);
}
